// UR_LSTM_24283745091977
// MI455X (gfx1250) — matrix operand order repaired and checked by reading, not compiled, not run on hardware
//
#include <hip/hip_runtime.h>

typedef __attribute__((ext_vector_type(16))) __bf16 v16bf;
typedef __attribute__((ext_vector_type(8)))  __bf16 v8bf;
typedef __attribute__((ext_vector_type(8)))  float  v8f;
typedef __attribute__((ext_vector_type(4)))  int    v4i;

// Problem constants
constexpr int Bb = 64;
constexpr int Tt = 512;
constexpr int Ii = 1024;
constexpr int Hh = 1024;
constexpr int G4 = 4 * Hh;       // 4096
constexpr int LDA_X = Tt * Ii;   // row stride of x in elements
constexpr int KTOT = Hh + Ii;    // 2048 (concat K: [h | x_t])
constexpr int KCH  = 64;         // K-chunk staged in LDS per double-buffer slot
constexpr int NCH  = KTOT / KCH; // 32 chunks
constexpr int CHE  = 64 * KCH;   // elements per chunk buffer (64 rows x 64 halves)

// ---------------------------------------------------------------------------
// CDNA5 async global->LDS copy (16B per lane, tracked by ASYNCcnt)
// ---------------------------------------------------------------------------
__device__ __forceinline__ void async_cp16(__bf16* lds_dst, const __bf16* gsrc) {
#if defined(__AMDGCN__)
#if __has_builtin(__builtin_amdgcn_global_load_async_to_lds_b128)
    __builtin_amdgcn_global_load_async_to_lds_b128(
        (__attribute__((address_space(1))) v4i*)gsrc,
        (__attribute__((address_space(3))) v4i*)lds_dst,
        0, 0);
#else
    unsigned loff = (unsigned)(uintptr_t)
        (__attribute__((address_space(3))) char*)(void*)lds_dst;
    asm volatile("global_load_async_to_lds_b128 %0, %1, off"
                 :: "v"(loff), "v"(gsrc) : "memory");
#endif
#endif
}

__device__ __forceinline__ void wait_async_prev() {  // <=4 outstanding
#if defined(__AMDGCN__)
#if __has_builtin(__builtin_amdgcn_s_wait_asynccnt)
    __builtin_amdgcn_s_wait_asynccnt(4);
#else
    asm volatile("s_wait_asynccnt 0x4" ::: "memory");
#endif
#endif
}

__device__ __forceinline__ void wait_async_all() {   // all complete
#if defined(__AMDGCN__)
#if __has_builtin(__builtin_amdgcn_s_wait_asynccnt)
    __builtin_amdgcn_s_wait_asynccnt(0);
#else
    asm volatile("s_wait_asynccnt 0x0" ::: "memory");
#endif
#endif
}

// ---------------------------------------------------------------------------
// fp32 -> bf16 elementwise convert
// ---------------------------------------------------------------------------
__global__ void cvt_f32_bf16(const float* __restrict__ src,
                             __bf16* __restrict__ dst, int n) {
    int i = blockIdx.x * blockDim.x + threadIdx.x;
    if (i < n) dst[i] = (__bf16)src[i];
}

// ---------------------------------------------------------------------------
// Per-timestep GEMM:  lin[64,4096] = [h | x_t] (64x2048 bf16) x Wcat^T (bf16)
// Wcat rows: [4096,1024] w_hh for K=0..1023, [4096,1024] w_ih for K=1024..2047
// Block: 256 threads = 8 waves. Block tile: M=64, N=64 (grid = 4096/64 = 64).
// Wave tile: M=16, N=32 (two v_wmma_f32_16x16x32_bf16 accumulators).
// A and B K-chunks (64 wide) staged in LDS via async-to-LDS, double buffered.
// ---------------------------------------------------------------------------
__global__ __launch_bounds__(256) void step_gemm(
    const __bf16* __restrict__ hbf,   // [64,1024] row-major
    const __bf16* __restrict__ xbf,   // pre-offset by t*I; row stride T*I
    const __bf16* __restrict__ whh,   // [4096,1024] row-major
    const __bf16* __restrict__ wih,   // [4096,1024] row-major
    float* __restrict__ lin)          // [64,4096]
{
    __shared__ __align__(32) __bf16 sA[2 * CHE];   // 16 KB
    __shared__ __align__(32) __bf16 sB[2 * CHE];   // 16 KB

    const int tid  = threadIdx.x;
    const int lane = tid & 31;
    const int wid  = tid >> 5;
    const int wm   = wid & 3;          // 0..3 -> M tile (16 rows)
    const int wn   = wid >> 2;         // 0..1 -> N half (32 cols)
    const int m0   = wm * 16;
    const int n0b  = blockIdx.x * 64;  // block's first output column
    const int nl   = lane & 15;
    const int hi   = lane >> 4;        // 0 or 1

    // Issue the async copy of chunk `c` (K range [kc, kc+64)) into buffer bi.
    // 512 x 16B segments each for A and B; each of 256 threads copies 2 + 2.
    auto issue_chunk = [&](int kc, int bi) {
        const bool ph0 = (kc < Hh);
        const int  kk  = ph0 ? kc : kc - Hh;
        const __bf16* W = ph0 ? whh : wih;
#pragma unroll
        for (int rep = 0; rep < 2; ++rep) {
            int s   = tid + rep * 256;      // 0..511
            int row = s >> 3;               // 0..63
            int col = (s & 7) * 8;          // half offset within chunk row
            const __bf16* asrc = ph0
                ? (hbf + (size_t)row * Hh + kk + col)
                : (xbf + (size_t)row * LDA_X + kk + col);
            async_cp16(sA + bi * CHE + row * KCH + col, asrc);
            const __bf16* bsrc = W + (size_t)(n0b + row) * Hh + kk + col;
            async_cp16(sB + bi * CHE + row * KCH + col, bsrc);
        }
    };

    v8f acc0 = {};
    v8f acc1 = {};

    issue_chunk(0, 0);

    for (int c = 0; c < NCH; ++c) {
        if (c + 1 < NCH) {
            issue_chunk((c + 1) * KCH, (c + 1) & 1);
            wait_async_prev();   // retire chunk c's 4 copies (in-order)
        } else {
            wait_async_all();
        }
        __syncthreads();         // chunk c visible to every wave

        const __bf16* As = sA + (c & 1) * CHE;
        const __bf16* Bs = sB + (c & 1) * CHE;
#pragma unroll
        for (int k = 0; k < KCH; k += 32) {
            v16bf a  = *(const v16bf*)(As + (m0 + nl) * KCH + k + hi * 16);
            v16bf b0 = *(const v16bf*)(Bs + (wn * 32 + nl) * KCH + k + hi * 16);
            v16bf b1 = *(const v16bf*)(Bs + (wn * 32 + 16 + nl) * KCH + k + hi * 16);
            acc0 = __builtin_amdgcn_wmma_f32_16x16x32_bf16(
                       false, a, false, b0, (short)0, acc0, false, false);
            acc1 = __builtin_amdgcn_wmma_f32_16x16x32_bf16(
                       false, a, false, b1, (short)0, acc1, false, false);
        }
        __syncthreads();         // reads done before buffer is re-filled
    }

    // ---- Store D (f32 16x16 C/D layout: row = 8*hi + vgpr, col = lane&15) ----
    float* op = lin + (size_t)(m0 + hi * 8) * G4 + n0b + wn * 32 + nl;
#pragma unroll
    for (int r = 0; r < 8; ++r) {
        op[(size_t)r * G4]      = acc0[r];
        op[(size_t)r * G4 + 16] = acc1[r];
    }
}

// ---------------------------------------------------------------------------
// Gate nonlinearity + state update (UR-LSTM)
// lin layout [64,4096], gate chunks of 1024: r, f, u, o
// ---------------------------------------------------------------------------
__device__ __forceinline__ float sigmoidf_(float x) {
    return 1.0f / (1.0f + __expf(-x));
}

__global__ __launch_bounds__(256) void step_gate(
    const float* __restrict__ lin,
    const float* __restrict__ b_ih,   // [4096]
    const float* __restrict__ b_hh,   // [4096]
    const float* __restrict__ fb,     // [1024] forget_bias
    float* __restrict__ c_ws,         // [64,1024] fp32 cell state
    __bf16* __restrict__ h_bf,        // [64,1024] bf16 hidden (next step's A)
    float* __restrict__ out,          // [B,T,H] fp32
    int t)
{
    int idx = blockIdx.x * blockDim.x + threadIdx.x;   // 0 .. 65535
    int b = idx >> 10;
    int j = idx & 1023;
    const float* lb = lin + (size_t)b * G4;

    float fbv = fb[j];
    float rl = lb[j]          + b_ih[j]          + b_hh[j]          - fbv;
    float fl = lb[Hh + j]     + b_ih[Hh + j]     + b_hh[Hh + j]     + fbv;
    float ul = lb[2 * Hh + j] + b_ih[2 * Hh + j] + b_hh[2 * Hh + j];
    float ol = lb[3 * Hh + j] + b_ih[3 * Hh + j] + b_hh[3 * Hh + j];

    float r = sigmoidf_(rl);
    float f = sigmoidf_(fl);
    float u = tanhf(ul);
    float o = sigmoidf_(ol);

    float fm = 1.0f - f;
    float g  = r * (1.0f - fm * fm) + (1.0f - r) * f * f;

    float c  = g * c_ws[idx] + (1.0f - g) * u;
    float h  = o * tanhf(c);

    c_ws[idx] = c;
    h_bf[idx] = (__bf16)h;
    out[(size_t)b * Tt * Hh + (size_t)t * Hh + j] = h;
}

// ---------------------------------------------------------------------------
// Emit final h (=out[:,T-1,:]) and c to the tail of d_out
// ---------------------------------------------------------------------------
__global__ void tail_copy(const float* __restrict__ c_ws,
                          float* __restrict__ out) {
    int idx = blockIdx.x * blockDim.x + threadIdx.x;  // 0 .. 65535
    int b = idx >> 10;
    int j = idx & 1023;
    float h = out[(size_t)b * Tt * Hh + (size_t)(Tt - 1) * Hh + j];
    size_t base = (size_t)Bb * Tt * Hh;
    out[base + idx]           = h;         // h_t [1,B,H]
    out[base + Bb * Hh + idx] = c_ws[idx]; // c_t [1,B,H]
}

// ---------------------------------------------------------------------------
// Host launcher
// ---------------------------------------------------------------------------
extern "C" void kernel_launch(void* const* d_in, const int* in_sizes, int n_in,
                              void* d_out, int out_size, void* d_ws, size_t ws_size,
                              hipStream_t stream) {
    const float* x   = (const float*)d_in[0];   // [B,T,I]
    const float* wih = (const float*)d_in[1];   // [4H,I]
    const float* bih = (const float*)d_in[2];   // [4H]
    const float* whh = (const float*)d_in[3];   // [4H,H]
    const float* bhh = (const float*)d_in[4];   // [4H]
    const float* fb  = (const float*)d_in[5];   // [H]
    const float* h0  = (const float*)d_in[6];   // [B,H]
    const float* c0  = (const float*)d_in[7];   // [B,H]
    float* out = (float*)d_out;

    // Workspace carve-up (all chunk sizes 256B-aligned)
    char* p = (char*)d_ws;
    __bf16* x_bf   = (__bf16*)p; p += (size_t)Bb * Tt * Ii * 2;  // 64 MB
    __bf16* wih_bf = (__bf16*)p; p += (size_t)G4 * Ii * 2;       // 8 MB
    __bf16* whh_bf = (__bf16*)p; p += (size_t)G4 * Hh * 2;       // 8 MB
    __bf16* h_bf   = (__bf16*)p; p += (size_t)Bb * Hh * 2;       // 128 KB
    float*  c_ws   = (float*)p;  p += (size_t)Bb * Hh * 4;       // 256 KB
    float*  lin    = (float*)p;                                  // 1 MB

    // One-time converts (per call; deterministic)
    {
        int n = Bb * Tt * Ii;
        cvt_f32_bf16<<<(n + 255) / 256, 256, 0, stream>>>(x, x_bf, n);
        n = G4 * Ii;
        cvt_f32_bf16<<<(n + 255) / 256, 256, 0, stream>>>(wih, wih_bf, n);
        n = G4 * Hh;
        cvt_f32_bf16<<<(n + 255) / 256, 256, 0, stream>>>(whh, whh_bf, n);
        n = Bb * Hh;
        cvt_f32_bf16<<<(n + 255) / 256, 256, 0, stream>>>(h0, h_bf, n);
    }
    (void)hipMemcpyAsync(c_ws, c0, (size_t)Bb * Hh * sizeof(float),
                         hipMemcpyDeviceToDevice, stream);

    const int gate_blocks = (Bb * Hh) / 256;  // 256
    for (int t = 0; t < Tt; ++t) {
        step_gemm<<<G4 / 64, 256, 0, stream>>>(
            h_bf, x_bf + (size_t)t * Ii, whh_bf, wih_bf, lin);
        step_gate<<<gate_blocks, 256, 0, stream>>>(
            lin, bih, bhh, fb, c_ws, h_bf, out, t);
    }
    tail_copy<<<gate_blocks, 256, 0, stream>>>(c_ws, out);
}
